// DecoderMultiHeadAttention_68642167324731
// MI455X (gfx1250) — matrix operand order repaired and checked by reading, not compiled, not run on hardware
//
#include <hip/hip_runtime.h>
#include <hip/hip_bf16.h>

typedef __attribute__((ext_vector_type(16))) _Float16 v16h;
typedef __attribute__((ext_vector_type(8)))  _Float16 v8h;
typedef __attribute__((ext_vector_type(8)))  float    v8f;
typedef __attribute__((ext_vector_type(4)))  unsigned int u32x4;
typedef __attribute__((ext_vector_type(8)))  int i32x8;
typedef __attribute__((ext_vector_type(4)))  int i32x4;

#define Bc 4
#define Tc 1024
#define Sc 1024
#define Dc 1024
#define Hc 16
#define DHc 64
#define BTc (Bc * Tc)
#define BSc (Bc * Sc)

#define KB   32          // K per WMMA step
#define LP   40          // LDS row pitch in halves (32 + 4 dwords pad; 80B, 16B aligned)

// ---------------------------------------------------------------------------
// CDNA5 feature probes (compile-safe: fall back to plain paths if missing)
// ---------------------------------------------------------------------------
#if __has_builtin(__builtin_amdgcn_tensor_load_to_lds)
#define HAVE_TDM 1
#if __has_include(<hip/amd_detail/amd_gfx1250_TDM.h>)
#define TDM_SIX_ARGS 1   // therock/clang-23 headers -> 6-arg builtin
#endif
#endif

#if __has_builtin(__builtin_amdgcn_global_load_async_to_lds_b128)
#define HAVE_ASYNC 1
#endif

__device__ __forceinline__ void wait_async0() {
#if __has_builtin(__builtin_amdgcn_s_wait_asynccnt)
  __builtin_amdgcn_s_wait_asynccnt(0);
#else
  asm volatile("s_wait_asynccnt 0x0" ::: "memory");
#endif
}

#if defined(HAVE_ASYNC)
__device__ __forceinline__ void async_copy_b128(const _Float16* gsrc, _Float16* lds_dst) {
  // Param types per clang diagnostic: (global v4i*, local v4i*, imm offset, imm cpol)
  __builtin_amdgcn_global_load_async_to_lds_b128(
      (__attribute__((address_space(1))) i32x4*)gsrc,
      (__attribute__((address_space(3))) i32x4*)lds_dst,
      0, 0);
}
#endif

#if defined(HAVE_TDM)
// TDM: stage a [rows x KB] f16 tile (row stride = strideElems) into LDS at
// lds_off, padding each 16-DWORD row with 4 DWORDs -> LDS pitch LP halves.
__device__ __forceinline__ void tdm_load_tile(const _Float16* g, unsigned lds_off,
                                              int rows, int strideElems) {
  unsigned long long ga = (unsigned long long)(size_t)g;
  u32x4 g0 = { 1u /*count=1*/, lds_off,
               (unsigned)ga, (unsigned)(ga >> 32) | (2u << 30) /*type=2*/ };
  unsigned td0 = (unsigned)strideElems;   // tensor_dim0 (>= tile_dim0; never OOB here)
  unsigned td1 = 0x7FFFFFu;               // tensor_dim1 (large; never OOB here)
  unsigned tile0 = (unsigned)KB;
  unsigned tile1 = (unsigned)rows;
  unsigned long long st0 = (unsigned long long)strideElems;
  // data_size=1 (2B) | pad_enable | pad_interval=3 (16 DW) | pad_amount=3 (4 DW)
  i32x8 g1 = {
    (int)((1u << 16) | (1u << 20) | (3u << 22) | (3u << 25)),
    (int)((td0 & 0xFFFFu) << 16),
    (int)(((td0 >> 16) & 0xFFFFu) | ((td1 & 0xFFFFu) << 16)),
    (int)(((td1 >> 16) & 0xFFFFu) | ((tile0 & 0xFFFFu) << 16)),
    (int)(tile1 & 0xFFFFu),
    (int)(unsigned)(st0 & 0xFFFFFFFFull),
    (int)(unsigned)((st0 >> 32) & 0xFFFFull),
    0 };
  i32x4 z4 = { 0, 0, 0, 0 };
#if defined(TDM_SIX_ARGS)
  i32x8 z8 = { 0, 0, 0, 0, 0, 0, 0, 0 };
  __builtin_amdgcn_tensor_load_to_lds(g0, g1, z4, z4, z8, 0);
#else
  __builtin_amdgcn_tensor_load_to_lds(g0, g1, z4, z4, 0);
#endif
}

__device__ __forceinline__ unsigned lds_offset_of(const void* p) {
  return (unsigned)(size_t)(__attribute__((address_space(3))) const char*)(const char*)p;
}
#endif

// ---------------------------------------------------------------------------
// WMMA helpers (CDNA5 / gfx1250, wave32)
// ---------------------------------------------------------------------------
__device__ __forceinline__ v8f wmma_f16(v16h a, v16h b, v8f c) {
  return __builtin_amdgcn_wmma_f32_16x16x32_f16(
      false, a, false, b, (short)0, c, false, false);
}

// A fragment: lane L holds row M=row+(L%16);
// K offsets: {8*(L>=16)+[0..7], 16+8*(L>=16)+[0..7]}
__device__ __forceinline__ v16h load_a_frag(const _Float16* lds, int row) {
  int lane = threadIdx.x & 31;
  int half = lane >> 4;
  const _Float16* p = lds + (size_t)(row + (lane & 15)) * LP;
  union { v16h v; v8h h[2]; } f;
  f.h[0] = *(const v8h*)(p + 8 * half);
  f.h[1] = *(const v8h*)(p + 16 + 8 * half);
  return f.v;
}

// B fragment from K-transposed LDS tile ldsT[n][k]:
// lane L holds col N=col+(L%16); K offsets: {8*(L>=16)+[0..7], 16+8*(L>=16)+[0..7]}
__device__ __forceinline__ v16h load_b_frag(const _Float16* ldsT, int col) {
  int lane = threadIdx.x & 31;
  int half = lane >> 4;
  const _Float16* p = ldsT + (size_t)(col + (lane & 15)) * LP;
  union { v16h v; v8h h[2]; } f;
  f.h[0] = *(const v8h*)(p + 8 * half);
  f.h[1] = *(const v8h*)(p + 16 + 8 * half);
  return f.v;
}

// ---------------------------------------------------------------------------
// Elementwise f32 -> f16 convert
// ---------------------------------------------------------------------------
__global__ __launch_bounds__(256) void cvt_f32_to_f16(const float* __restrict__ s,
                                                      _Float16* __restrict__ d,
                                                      size_t n) {
  size_t i = ((size_t)blockIdx.x * 256 + threadIdx.x) * 4;
  if (i + 3 < n) {
    float4 v = *(const float4*)(s + i);
    d[i + 0] = (_Float16)v.x;
    d[i + 1] = (_Float16)v.y;
    d[i + 2] = (_Float16)v.z;
    d[i + 3] = (_Float16)v.w;
  }
}

// ---------------------------------------------------------------------------
// Block reductions
// ---------------------------------------------------------------------------
__device__ __forceinline__ float block_sum(float v, float* sh) {
  int tid = threadIdx.x;
  sh[tid] = v; __syncthreads();
  for (int s = 128; s > 0; s >>= 1) {
    if (tid < s) sh[tid] += sh[tid + s];
    __syncthreads();
  }
  float r = sh[0]; __syncthreads();
  return r;
}

__device__ __forceinline__ float block_max(float v, float* sh) {
  int tid = threadIdx.x;
  sh[tid] = v; __syncthreads();
  for (int s = 128; s > 0; s >>= 1) {
    if (tid < s) sh[tid] = fmaxf(sh[tid], sh[tid + s]);
    __syncthreads();
  }
  float r = sh[0]; __syncthreads();
  return r;
}

// ---------------------------------------------------------------------------
// LayerNorm (per row of D=1024) with f16 output
// ---------------------------------------------------------------------------
__global__ __launch_bounds__(256) void layernorm_f16(const float* __restrict__ x,
                                                     const float* __restrict__ g,
                                                     const float* __restrict__ b,
                                                     _Float16* __restrict__ out) {
  __shared__ float sh[256];
  int row = blockIdx.x;
  int tid = threadIdx.x;
  const float* xr = x + (size_t)row * Dc;

  float s = 0.f;
  for (int i = tid; i < Dc; i += 256) s += xr[i];
  float mean = block_sum(s, sh) * (1.0f / Dc);

  float v = 0.f;
  for (int i = tid; i < Dc; i += 256) {
    float d = xr[i] - mean;
    v += d * d;
  }
  float var = block_sum(v, sh) * (1.0f / Dc);
  float inv = rsqrtf(var + 1e-5f);

  _Float16* o = out + (size_t)row * Dc;
  for (int i = tid; i < Dc; i += 256)
    o[i] = (_Float16)((xr[i] - mean) * inv * g[i] + b[i]);
}

// ---------------------------------------------------------------------------
// Projection GEMM: C[M,N] = A_f16[M,K] @ W_f16[K,N] + bias
// WG tile 64x256, 8 waves, each wave 32x64 (2x4 fragments), K step 32.
//   out16 != null : store f16;  else acc32 += result (residual accumulate)
// A-tile staged by TDM (tensor_load_to_lds + s_wait_tensorcnt) when available.
// ---------------------------------------------------------------------------
__global__ __launch_bounds__(256) void gemm_proj(const _Float16* __restrict__ A,
                                                 const _Float16* __restrict__ W,
                                                 const float* __restrict__ bias,
                                                 _Float16* __restrict__ out16,
                                                 float* __restrict__ acc32,
                                                 int M, int N, int K) {
  __shared__ _Float16 sA[64][LP];
  __shared__ _Float16 sW[256][LP];  // K-transposed: sW[n][k]

  int tid  = threadIdx.x;
  int wave = tid >> 5;
  int lane = tid & 31;
  int n0 = blockIdx.x * 256;
  int m0 = blockIdx.y * 64;
  int wm = (wave >> 2) * 32;  // 0 or 32
  int wn = (wave & 3) * 64;   // 0,64,128,192

  v8f c[2][4] = {};

  for (int k0 = 0; k0 < K; k0 += KB) {
    // --- stage A tile: 64 rows x 32 halves (TDM if available)
#if defined(HAVE_TDM)
    if (tid < 32) {
      tdm_load_tile(A + (size_t)m0 * K + k0, lds_offset_of(&sA[0][0]), 64, K);
      __builtin_amdgcn_s_wait_tensorcnt(0);
    }
#else
    {
      int r  = tid >> 2;
      int cg = (tid & 3) * 8;
      const _Float16* src = A + (size_t)(m0 + r) * K + k0 + cg;
      if (k0 + KB < K) __builtin_prefetch(src + KB, 0, 3);
      *(v8h*)&sA[r][cg] = *(const v8h*)src;
    }
#endif
    // --- stage W tile K-transposed: 32 k-rows x 256 n-cols
    {
      int kk = tid >> 3;
      int ng = (tid & 7) * 32;
      const _Float16* src = W + (size_t)(k0 + kk) * N + n0 + ng;
      if (k0 + KB < K) __builtin_prefetch(src + (size_t)KB * N, 0, 3);
#pragma unroll
      for (int q = 0; q < 4; ++q) {
        v8h v0 = *(const v8h*)(src + q * 8);
#pragma unroll
        for (int j = 0; j < 8; ++j) sW[ng + q * 8 + j][kk] = v0[j];
      }
    }
    __syncthreads();

    v16h a0 = load_a_frag(&sA[0][0], wm);
    v16h a1 = load_a_frag(&sA[0][0], wm + 16);
#pragma unroll
    for (int fn = 0; fn < 4; ++fn) {
      v16h bfrag = load_b_frag(&sW[0][0], wn + fn * 16);
      c[0][fn] = wmma_f16(a0, bfrag, c[0][fn]);
      c[1][fn] = wmma_f16(a1, bfrag, c[1][fn]);
    }
    __syncthreads();
  }

  int half = lane >> 4;
  int nl   = lane & 15;
#pragma unroll
  for (int fm = 0; fm < 2; ++fm)
#pragma unroll
    for (int fn = 0; fn < 4; ++fn) {
      int col = n0 + wn + fn * 16 + nl;
      float bv = bias[col];
#pragma unroll
      for (int r = 0; r < 8; ++r) {
        int row = m0 + wm + fm * 16 + r + half * 8;
        float val = c[fm][fn][r] + bv;
        if (out16) out16[(size_t)row * N + col] = (_Float16)val;
        else       acc32[(size_t)row * N + col] += val;
      }
    }
}

// ---------------------------------------------------------------------------
// Attention scores: sc[bh][qi][ki] = mask ? (Q.K^T)*0.125 : -1e30
// Q: [B,Lq,D] f16 (head h at cols h*64), Km: [B,Lk,D] f16. K-dim = 64.
// Grid: (Lk/128, Lq/64, B*H). Tiles staged with async global->LDS loads.
// ---------------------------------------------------------------------------
__global__ __launch_bounds__(256) void attn_scores(const _Float16* __restrict__ Q,
                                                   const _Float16* __restrict__ Km,
                                                   float* __restrict__ sc,
                                                   const int* __restrict__ lens,
                                                   int causal, int Lq, int Lk) {
  __shared__ _Float16 sA[64][LP];
  __shared__ _Float16 sW[128][LP];  // sW[n][k] = Km[b, n][h*64+k] (naturally transposed)

  int tid  = threadIdx.x;
  int wave = tid >> 5;
  int lane = tid & 31;
  int bh = blockIdx.z;
  int b  = bh >> 4;
  int h  = bh & 15;
  int n0 = blockIdx.x * 128;
  int m0 = blockIdx.y * 64;
  int wm = (wave >> 2) * 32;
  int wn = (wave & 3) * 32;

  const _Float16* Qb = Q  + (size_t)b * Lq * Dc + h * DHc;
  const _Float16* Kb = Km + (size_t)b * Lk * Dc + h * DHc;

  v8f c[2][2] = {};

  for (int k0 = 0; k0 < DHc; k0 += KB) {
    {
      int r  = tid >> 2;
      int cg = (tid & 3) * 8;
      const _Float16* src = Qb + (size_t)(m0 + r) * Dc + k0 + cg;
#if defined(HAVE_ASYNC)
      async_copy_b128(src, &sA[r][cg]);
#else
      *(v8h*)&sA[r][cg] = *(const v8h*)src;
#endif
    }
    {
      int n  = tid >> 1;
      int cg = (tid & 1) * 16;
      const _Float16* src = Kb + (size_t)(n0 + n) * Dc + k0 + cg;
#if defined(HAVE_ASYNC)
      async_copy_b128(src,     &sW[n][cg]);
      async_copy_b128(src + 8, &sW[n][cg + 8]);
#else
      *(v8h*)&sW[n][cg]     = *(const v8h*)(src);
      *(v8h*)&sW[n][cg + 8] = *(const v8h*)(src + 8);
#endif
    }
#if defined(HAVE_ASYNC)
    wait_async0();
#endif
    __syncthreads();

    v16h a0 = load_a_frag(&sA[0][0], wm);
    v16h a1 = load_a_frag(&sA[0][0], wm + 16);
    v16h b0 = load_b_frag(&sW[0][0], wn);
    v16h b1 = load_b_frag(&sW[0][0], wn + 16);
    c[0][0] = wmma_f16(a0, b0, c[0][0]);
    c[0][1] = wmma_f16(a0, b1, c[0][1]);
    c[1][0] = wmma_f16(a1, b0, c[1][0]);
    c[1][1] = wmma_f16(a1, b1, c[1][1]);
    __syncthreads();
  }

  int half = lane >> 4;
  int nl   = lane & 15;
  int len  = lens[b];
  float* scb = sc + (size_t)bh * Lq * Lk;
#pragma unroll
  for (int fm = 0; fm < 2; ++fm)
#pragma unroll
    for (int fn = 0; fn < 2; ++fn) {
#pragma unroll
      for (int r = 0; r < 8; ++r) {
        int qi = m0 + wm + fm * 16 + r + half * 8;
        int ki = n0 + wn + fn * 16 + nl;
        float s = c[fm][fn][r] * 0.125f;  // DH^-0.5
        bool ok = causal ? (ki <= qi && ki < len) : (ki < len);
        scb[(size_t)qi * Lk + ki] = ok ? s : -1e30f;
      }
    }
}

// ---------------------------------------------------------------------------
// Row softmax in place over Lk elements; one block per row.
// ---------------------------------------------------------------------------
__global__ __launch_bounds__(256) void softmax_row(float* __restrict__ sc, int Lk) {
  __shared__ float sh[256];
  int tid = threadIdx.x;
  float* p = sc + (size_t)blockIdx.x * Lk;

  float mx = -3.0e38f;
  for (int i = tid; i < Lk; i += 256) mx = fmaxf(mx, p[i]);
  mx = block_max(mx, sh);

  float sum = 0.f;
  for (int i = tid; i < Lk; i += 256) {
    float e = __expf(p[i] - mx);
    p[i] = e;
    sum += e;
  }
  sum = block_sum(sum, sh);
  float inv = 1.0f / sum;
  for (int i = tid; i < Lk; i += 256) p[i] *= inv;
}

// ---------------------------------------------------------------------------
// Context GEMM: ctx[b,q,h*64+d] = sum_k P[bh][q][k] * V[b,k,h*64+d]
// A = probs f32 (converted to f16 in LDS), B = V f16. M=Lq, N=64, K=Lk.
// Grid: (Lq/64, B*H). WG tile 64x64; 8 waves, each 16x32.
// ---------------------------------------------------------------------------
__global__ __launch_bounds__(256) void attn_ctx(const float* __restrict__ P,
                                                const _Float16* __restrict__ V,
                                                _Float16* __restrict__ ctx,
                                                int Lq, int Lk) {
  __shared__ _Float16 sA[64][LP];
  __shared__ _Float16 sW[64][LP];  // sW[n][k] = V[b, k][h*64+n]

  int tid  = threadIdx.x;
  int wave = tid >> 5;
  int lane = tid & 31;
  int bh = blockIdx.y;
  int b  = bh >> 4;
  int h  = bh & 15;
  int m0 = blockIdx.x * 64;
  int wm = (wave >> 1) * 16;
  int wn = (wave & 1) * 32;

  const float*    Pb = P + (size_t)bh * Lq * Lk;
  const _Float16* Vb = V + (size_t)b * Lk * Dc + h * DHc;

  v8f c[2] = {};

  for (int k0 = 0; k0 < Lk; k0 += KB) {
    {
      int r  = tid >> 2;
      int cg = (tid & 3) * 8;
      const float* src = Pb + (size_t)(m0 + r) * Lk + k0 + cg;
#pragma unroll
      for (int j = 0; j < 8; ++j) sA[r][cg + j] = (_Float16)src[j];
    }
    {
      int kk = tid >> 3;
      int ng = (tid & 7) * 8;
      v8h v0 = *(const v8h*)(Vb + (size_t)(k0 + kk) * Dc + ng);
#pragma unroll
      for (int j = 0; j < 8; ++j) sW[ng + j][kk] = v0[j];
    }
    __syncthreads();

    v16h a  = load_a_frag(&sA[0][0], wm);
    v16h b0 = load_b_frag(&sW[0][0], wn);
    v16h b1 = load_b_frag(&sW[0][0], wn + 16);
    c[0] = wmma_f16(a, b0, c[0]);
    c[1] = wmma_f16(a, b1, c[1]);
    __syncthreads();
  }

  int half = lane >> 4;
  int nl   = lane & 15;
#pragma unroll
  for (int fn = 0; fn < 2; ++fn) {
    int col = wn + fn * 16 + nl;
#pragma unroll
    for (int r = 0; r < 8; ++r) {
      int q = m0 + wm + r + half * 8;
      ctx[((size_t)b * Lq + q) * Dc + h * DHc + col] = (_Float16)c[fn][r];
    }
  }
}

// ---------------------------------------------------------------------------
// Host-side orchestration
// ---------------------------------------------------------------------------
extern "C" void kernel_launch(void* const* d_in, const int* in_sizes, int n_in,
                              void* d_out, int out_size, void* d_ws, size_t ws_size,
                              hipStream_t stream) {
  (void)in_sizes; (void)n_in; (void)out_size; (void)ws_size;

  const float* Wf[8];
  const float* bf[8];
  for (int i = 0; i < 8; ++i) {
    Wf[i] = (const float*)d_in[2 * i];
    bf[i] = (const float*)d_in[2 * i + 1];
  }
  const float* x       = (const float*)d_in[16];
  const float* enc     = (const float*)d_in[17];
  const int*   src_len = (const int*)d_in[18];
  const int*   tgt_len = (const int*)d_in[19];
  const float* ln1g    = (const float*)d_in[20];
  const float* ln1b    = (const float*)d_in[21];
  const float* ln2g    = (const float*)d_in[22];
  const float* ln2b    = (const float*)d_in[23];

  // Workspace layout (f16): 8 weight matrices, xn, enc, Q, K, V, ctx (~64 MB)
  _Float16* wsh = (_Float16*)d_ws;
  _Float16* w16[8];
  for (int i = 0; i < 8; ++i) w16[i] = wsh + (size_t)i * Dc * Dc;
  _Float16* xn16  = wsh + (size_t)8 * Dc * Dc;
  _Float16* enc16 = xn16 + (size_t)BTc * Dc;
  _Float16* Q16   = enc16 + (size_t)BSc * Dc;
  _Float16* K16   = Q16 + (size_t)BTc * Dc;
  _Float16* V16   = K16 + (size_t)BSc * Dc;
  _Float16* ctx16 = V16 + (size_t)BSc * Dc;

  float* xout = (float*)d_out;                 // [B,T,D] running residual
  float* caw  = xout + (size_t)BTc * Dc;       // [B,H,T,S] (self-attn scratch, then output)

  dim3 blk(256);
  dim3 gGemm(Dc / 256, BTc / 64);
  dim3 gScores(Sc / 128, Tc / 64, Bc * Hc);
  dim3 gCtx(Tc / 64, Bc * Hc);

  for (int i = 0; i < 8; ++i)
    cvt_f32_to_f16<<<(Dc * Dc) / 1024, blk, 0, stream>>>(Wf[i], w16[i], (size_t)Dc * Dc);
  cvt_f32_to_f16<<<(BSc * Dc) / 1024, blk, 0, stream>>>(enc, enc16, (size_t)BSc * Dc);

  (void)hipMemcpyAsync(xout, x, (size_t)BTc * Dc * sizeof(float),
                       hipMemcpyDeviceToDevice, stream);

  // ---- Self-attention block (pre-LN) ----
  layernorm_f16<<<BTc, blk, 0, stream>>>(x, ln1g, ln1b, xn16);
  gemm_proj<<<gGemm, blk, 0, stream>>>(xn16, w16[0], bf[0], Q16, nullptr, BTc, Dc, Dc);
  gemm_proj<<<gGemm, blk, 0, stream>>>(xn16, w16[1], bf[1], K16, nullptr, BTc, Dc, Dc);
  gemm_proj<<<gGemm, blk, 0, stream>>>(xn16, w16[2], bf[2], V16, nullptr, BTc, Dc, Dc);
  attn_scores<<<gScores, blk, 0, stream>>>(Q16, K16, caw, tgt_len, 1, Tc, Tc);
  softmax_row<<<Bc * Hc * Tc, blk, 0, stream>>>(caw, Tc);
  attn_ctx<<<gCtx, blk, 0, stream>>>(caw, V16, ctx16, Tc, Tc);
  gemm_proj<<<gGemm, blk, 0, stream>>>(ctx16, w16[3], bf[3], nullptr, xout, BTc, Dc, Dc);

  // ---- Cross-attention block (pre-LN) ----
  layernorm_f16<<<BTc, blk, 0, stream>>>(xout, ln2g, ln2b, xn16);
  gemm_proj<<<gGemm, blk, 0, stream>>>(xn16,  w16[4], bf[4], Q16, nullptr, BTc, Dc, Dc);
  gemm_proj<<<gGemm, blk, 0, stream>>>(enc16, w16[5], bf[5], K16, nullptr, BSc, Dc, Dc);
  gemm_proj<<<gGemm, blk, 0, stream>>>(enc16, w16[6], bf[6], V16, nullptr, BSc, Dc, Dc);
  attn_scores<<<gScores, blk, 0, stream>>>(Q16, K16, caw, src_len, 0, Tc, Sc);
  softmax_row<<<Bc * Hc * Tc, blk, 0, stream>>>(caw, Sc);
  attn_ctx<<<gCtx, blk, 0, stream>>>(caw, V16, ctx16, Tc, Sc);
  gemm_proj<<<gGemm, blk, 0, stream>>>(ctx16, w16[7], bf[7], nullptr, xout, BTc, Dc, Dc);
}
